// FasterRCNN_1640677507309
// MI455X (gfx1250) — compile-verified, matrix operand order checked by reading, not run on hardware
//
#include <hip/hip_runtime.h>
#include <math.h>

// ---------------------------------------------------------------------------
// Faster-RCNN inference for gfx1250 (MI455X): bf16 WMMA GEMM everywhere the
// FLOPs live (backbone im2col convs, RPN convs, FC head), faithful device-side
// anchors / argsort / greedy-NMS / roi_pool for the glue.
// ---------------------------------------------------------------------------

typedef __bf16 bf16;
typedef __attribute__((ext_vector_type(16))) __bf16 bf16x16;
typedef __attribute__((ext_vector_type(8)))  float  v8f;

#define CHUNK_BYTES (32u << 20)   // im2col staging chunk
#define NANCH 3600
#define NKEEP 2000

// ------------------------------- WMMA GEMM ---------------------------------
// C[M,N] (f32, ldc) = A[M,K] (bf16, K-major, ld=K) * W[N,K]^T (bf16, K-major)
// K must be a multiple of 32 (callers pad). Row/col guards handle ragged M,N.

union FragU { bf16x16 v; float4 q[2]; };

__device__ __forceinline__ bf16x16 load_frag(const bf16* __restrict__ base,
                                             int ld, int row, int nrows, int k0) {
  int lane = threadIdx.x & 31;
  int r = row + (lane & 15);
  if (r >= nrows) r = nrows - 1;          // clamp; stores are guarded
  int kof = (lane & 16) ? 8 : 0;          // lanes 16-31 hold K 8..15 / 24..31
  const bf16* p = base + (long)r * ld + (k0 + kof);
  FragU f;
  f.q[0] = *reinterpret_cast<const float4*>(p);        // K k0+kof .. +7
  f.q[1] = *reinterpret_cast<const float4*>(p + 16);   // K k0+kof+16 .. +23
  return f.v;
}

__global__ void __launch_bounds__(256)
k_gemm_bf16(const bf16* __restrict__ A, const bf16* __restrict__ W,
            float* __restrict__ C, int M, int N, int K, int ldc,
            const float* __restrict__ bias, int relu) {
  int wave = threadIdx.x >> 5;
  int wm = wave >> 1;                    // 0..3
  int wn = wave & 1;                     // 0..1
  int tm = blockIdx.x * 128 + wm * 32;
  int tn = blockIdx.y * 64  + wn * 32;

  v8f a00 = {}, a01 = {}, a10 = {}, a11 = {};
  for (int k0 = 0; k0 < K; k0 += 32) {
    if (k0 + 32 < K) {
      __builtin_prefetch(A + (long)((tm < M ? tm : M - 1)) * K + k0 + 32, 0, 1);
      __builtin_prefetch(W + (long)((tn < N ? tn : N - 1)) * K + k0 + 32, 0, 1);
    }
    bf16x16 fa0 = load_frag(A, K, tm,      M, k0);
    bf16x16 fa1 = load_frag(A, K, tm + 16, M, k0);
    bf16x16 fb0 = load_frag(W, K, tn,      N, k0);
    bf16x16 fb1 = load_frag(W, K, tn + 16, N, k0);
    a00 = __builtin_amdgcn_wmma_f32_16x16x32_bf16(false, fa0, false, fb0, (short)0, a00, false, false);
    a01 = __builtin_amdgcn_wmma_f32_16x16x32_bf16(false, fa0, false, fb1, (short)0, a01, false, false);
    a10 = __builtin_amdgcn_wmma_f32_16x16x32_bf16(false, fa1, false, fb0, (short)0, a10, false, false);
    a11 = __builtin_amdgcn_wmma_f32_16x16x32_bf16(false, fa1, false, fb1, (short)0, a11, false, false);
  }

  int lane = threadIdx.x & 31;
  int cl = lane & 15;
  int rh = (lane & 16) ? 8 : 0;
  v8f accs[4] = {a00, a01, a10, a11};
#pragma unroll
  for (int t = 0; t < 4; ++t) {
    int mi = t >> 1, ni = t & 1;
#pragma unroll
    for (int j = 0; j < 8; ++j) {
      int row = tm + mi * 16 + j + rh;
      int col = tn + ni * 16 + cl;
      if (row < M && col < N) {
        float v = accs[t][j];
        if (bias) v += bias[col];
        if (relu) v = fmaxf(v, 0.f);
        C[(long)row * ldc + col] = v;
      }
    }
  }
}

// --------------------------- layout / packing ------------------------------

// conv weight OIHW f32 -> [O, Kpad] bf16, K ordered (ky, kx, ci)
__global__ void k_pack_conv_w(const float* __restrict__ w, bf16* __restrict__ dst,
                              int O, int I, int Kpad) {
  size_t idx = (size_t)blockIdx.x * blockDim.x + threadIdx.x;
  size_t tot = (size_t)O * Kpad;
  if (idx >= tot) return;
  int o = (int)(idx / Kpad), k = (int)(idx % Kpad);
  float v = 0.f;
  if (k < 9 * I) {
    int ky = k / (3 * I);
    int rem = k % (3 * I);
    int kx = rem / I;
    int ci = rem % I;
    v = w[(((size_t)o * I + ci) * 3 + ky) * 3 + kx];
  }
  dst[idx] = (bf16)v;
}

// linear weight [N,K] f32 -> [N,Kpad] bf16
__global__ void k_pack_lin_w(const float* __restrict__ w, bf16* __restrict__ dst,
                             int N, int K, int Kpad) {
  size_t idx = (size_t)blockIdx.x * blockDim.x + threadIdx.x;
  size_t tot = (size_t)N * Kpad;
  if (idx >= tot) return;
  int n = (int)(idx / Kpad), k = (int)(idx % Kpad);
  dst[idx] = (bf16)((k < K) ? w[(size_t)n * K + k] : 0.f);
}

__global__ void k_f32_to_bf16(const float* __restrict__ s, bf16* __restrict__ d, long n) {
  long i = (long)blockIdx.x * blockDim.x + threadIdx.x;
  if (i < n) d[i] = (bf16)s[i];
}

__global__ void k_nchw_to_nhwc(const float* __restrict__ src, float* __restrict__ dst,
                               int C, int H, int W) {
  long i = (long)blockIdx.x * blockDim.x + threadIdx.x;
  long tot = (long)C * H * W;
  if (i >= tot) return;
  int c = (int)(i % C);
  long p = i / C;
  int x = (int)(p % W);
  int y = (int)(p / W);
  dst[i] = src[((long)c * H + y) * W + x];
}

// 2x2 stride-2 maxpool, NHWC
__global__ void k_maxpool(const float* __restrict__ src, float* __restrict__ dst,
                          int H, int W, int C) {
  int HO = H >> 1, WO = W >> 1;
  long i = (long)blockIdx.x * blockDim.x + threadIdx.x;
  long tot = (long)HO * WO * C;
  if (i >= tot) return;
  int c = (int)(i % C);
  long p = i / C;
  int x = (int)(p % WO);
  int y = (int)(p / WO);
  const float* s0 = src + (((long)(2 * y) * W) + 2 * x) * C + c;
  const float* s1 = s0 + (long)W * C;
  dst[i] = fmaxf(fmaxf(s0[0], s0[C]), fmaxf(s1[0], s1[C]));
}

// NHWC f32 -> im2col rows [m0, m0+rows) of [Kpad] bf16, 3x3 SAME, (ky,kx,c)
__global__ void k_im2col(const float* __restrict__ src, bf16* __restrict__ A,
                         int H, int W, int C, int Kpad, int m0, int rows) {
  long idx = (long)blockIdx.x * blockDim.x + threadIdx.x;
  long tot = (long)rows * Kpad;
  if (idx >= tot) return;
  int r = (int)(idx / Kpad), k = (int)(idx % Kpad);
  int m = m0 + r;
  int y = m / W, x = m % W;
  float v = 0.f;
  if (k < 9 * C) {
    int ky = k / (3 * C);
    int rem = k % (3 * C);
    int kx = rem / C;
    int c = rem % C;
    int yy = y + ky - 1, xx = x + kx - 1;
    if (yy >= 0 && yy < H && xx >= 0 && xx < W)
      v = src[((long)yy * W + xx) * C + c];
  }
  A[(long)r * Kpad + k] = (bf16)v;
}

// ------------------------------ RPN glue -----------------------------------

__global__ void k_softmax2(const float* __restrict__ t, float* __restrict__ outp,
                           float* __restrict__ scores) {
  int q = blockIdx.x * blockDim.x + threadIdx.x;
  if (q >= NANCH) return;
  float a = t[q * 2], b = t[q * 2 + 1];
  float m = fmaxf(a, b);
  float ea = expf(a - m), eb = expf(b - m);
  float s = ea + eb;
  outp[q * 2] = ea / s;
  outp[q * 2 + 1] = eb / s;
  scores[q] = eb / s;
}

// anchors faithful to generate_anchor_mesh (incl. the quirky voc<->yolo swaps)
__global__ void k_anchors(float* __restrict__ anchors) {
  int idx = blockIdx.x * blockDim.x + threadIdx.x;
  if (idx >= NANCH) return;
  int a = idx % 9, p = idx / 9;
  int ridx = a / 3, sidx = a % 3;
  const float R[3] = {0.5f, 1.f, 2.f};
  const float S[3] = {8.f, 16.f, 32.f};
  const float stride = 16.f;                       // 320/20 (both axes)
  float b0 = 0.f, b1 = 0.f, b2 = stride - 1.f, b3 = stride - 1.f;
  float ya0 = b0 + 0.5f * (b3 - 1.f), ya1 = b1 + 0.5f * (b2 - 1.f);
  float ya2 = b3 - b1 + 1.f, ya3 = b2 - b0 + 1.f;
  float ratio = R[ridx];
  float wr = rintf(sqrtf(ya2 * ya3 / ratio));      // np.round == RNE
  float hr = rintf(wr * ratio);
  float r0 = ya0 - 0.5f * (wr - 1.f), r1 = ya1 - 0.5f * (hr - 1.f);
  float r2 = ya0 + 0.5f * (wr - 1.f), r3 = ya1 + 0.5f * (hr - 1.f);
  float yb0 = r0 + 0.5f * (r3 - 1.f), yb1 = r1 + 0.5f * (r2 - 1.f);
  float yb2 = r3 - r1 + 1.f, yb3 = r2 - r0 + 1.f;
  float w = yb2 * S[sidx], h = yb3 * S[sidx];
  float x1 = yb0 - 0.5f * (w - 1.f), y1 = yb1 - 0.5f * (h - 1.f);
  float x2 = yb0 + 0.5f * (w - 1.f), y2 = yb1 + 0.5f * (h - 1.f);
  float sx = (float)(p % 20) * stride, sy = (float)(p / 20) * stride;
  x1 += sx; x2 += sx; y1 += sy; y2 += sy;
  anchors[idx * 4 + 0] = fminf(fmaxf(x1, 0.f), 320.f);
  anchors[idx * 4 + 1] = fminf(fmaxf(y1, 0.f), 320.f);
  anchors[idx * 4 + 2] = fminf(fmaxf(x2, 0.f), 320.f);
  anchors[idx * 4 + 3] = fminf(fmaxf(y2, 0.f), 320.f);
}

// ascending (score, idx) bitonic sort of 3600 scores; emits top = anchors[order]
__global__ void k_sort_top(const float* __restrict__ scores,
                           const float* __restrict__ anchors,
                           float* __restrict__ topbox) {
  __shared__ float key[4096];
  __shared__ int   sid[4096];
  int tid = threadIdx.x, nt = blockDim.x;
  for (int i = tid; i < 4096; i += nt) {
    key[i] = (i < NANCH) ? scores[i] : INFINITY;
    sid[i] = (i < NANCH) ? i : (1 << 30);
  }
  __syncthreads();
  for (int k2 = 2; k2 <= 4096; k2 <<= 1) {
    for (int j = k2 >> 1; j > 0; j >>= 1) {
      for (int i = tid; i < 4096; i += nt) {
        int ixj = i ^ j;
        if (ixj > i) {
          bool up = ((i & k2) == 0);
          float ki = key[i], kj = key[ixj];
          int ii = sid[i], ij = sid[ixj];
          bool gt = (ki > kj) || (ki == kj && ii > ij);
          if (gt == up) { key[i] = kj; key[ixj] = ki; sid[i] = ij; sid[ixj] = ii; }
        }
      }
      __syncthreads();
    }
  }
  for (int i = tid; i < NANCH; i += nt) {
    int o = sid[i];
    topbox[i * 4 + 0] = anchors[o * 4 + 0];
    topbox[i * 4 + 1] = anchors[o * 4 + 1];
    topbox[i * 4 + 2] = anchors[o * 4 + 2];
    topbox[i * 4 + 3] = anchors[o * 4 + 3];
  }
}

// greedy NMS, score = y1 column (faithful), K=2000, argmax = lowest-index max
__global__ void k_nms(const float* __restrict__ topbox, int* __restrict__ keep) {
  __shared__ float sw[NANCH];
  __shared__ float areas[NANCH];
  __shared__ float red_v[256];
  __shared__ int   red_i[256];
  __shared__ float cb[5];
  __shared__ int   pick[2];   // pick[0]=idx, pick[1]=valid
  int tid = threadIdx.x;
  for (int i = tid; i < NANCH; i += 256) {
    float x1 = topbox[i * 4], y1 = topbox[i * 4 + 1];
    float x2 = topbox[i * 4 + 2], y2 = topbox[i * 4 + 3];
    areas[i] = (x2 - x1) * (y2 - y1);
    sw[i] = y1;
  }
  __syncthreads();
  for (int k = 0; k < NKEEP; ++k) {
    float bv = -INFINITY; int bi = 0x7fffffff;
    for (int i = tid; i < NANCH; i += 256) {
      float v = sw[i];
      if (v > bv || (v == bv && i < bi)) { bv = v; bi = i; }
    }
    red_v[tid] = bv; red_i[tid] = bi;
    __syncthreads();
    for (int s = 128; s > 0; s >>= 1) {
      if (tid < s) {
        float ov = red_v[tid + s]; int oi = red_i[tid + s];
        if (ov > red_v[tid] || (ov == red_v[tid] && oi < red_i[tid])) {
          red_v[tid] = ov; red_i[tid] = oi;
        }
      }
      __syncthreads();
    }
    if (tid == 0) {
      int i = red_i[0];
      float v = red_v[0];
      int vld = (v > -INFINITY) ? 1 : 0;  // isfinite (scores are coords or -inf)
      pick[0] = i; pick[1] = vld;
      keep[k] = vld ? i : -1;
      cb[0] = topbox[i * 4];     cb[1] = topbox[i * 4 + 1];
      cb[2] = topbox[i * 4 + 2]; cb[3] = topbox[i * 4 + 3];
      cb[4] = areas[i];
    }
    __syncthreads();
    if (pick[1]) {
      int i = pick[0];
      float ax1 = cb[0], ay1 = cb[1], ax2 = cb[2], ay2 = cb[3], ai = cb[4];
      for (int j = tid; j < NANCH; j += 256) {
        float xx1 = fmaxf(ax1, topbox[j * 4]);
        float yy1 = fmaxf(ay1, topbox[j * 4 + 1]);
        float xx2 = fminf(ax2, topbox[j * 4 + 2]);
        float yy2 = fminf(ay2, topbox[j * 4 + 3]);
        float inter = fmaxf(xx2 - xx1, 0.f) * fmaxf(yy2 - yy1, 0.f);
        float iou = inter / (ai + areas[j] - inter);
        if (iou > 0.6f || j == i) sw[j] = -INFINITY;
      }
    }
    __syncthreads();
  }
}

// rois = anchors[max(keep,0)]  (source's quirky re-index of original anchors)
__global__ void k_rois(const float* __restrict__ anchors, const int* __restrict__ keep,
                       float* __restrict__ rois, float* __restrict__ out_rois) {
  int r = blockIdx.x * blockDim.x + threadIdx.x;
  if (r >= NKEEP) return;
  int i = keep[r]; if (i < 0) i = 0;
#pragma unroll
  for (int d = 0; d < 4; ++d) {
    float v = anchors[i * 4 + d];
    rois[r * 4 + d] = v;
    out_rois[r * 4 + d] = v;
  }
}

// torchvision roi_pool semantics; feat NHWC [20,20,512]; out (c,py,px) bf16
__global__ void k_roi_pool(const float* __restrict__ feat, const float* __restrict__ rois,
                           bf16* __restrict__ pooled) {
  int b = blockIdx.x;           // 0 .. 2000*49-1
  int r = b / 49, pp = b % 49;
  int py = pp / 7, px = pp % 7;
  float x1 = rintf(rois[r * 4 + 0]), y1 = rintf(rois[r * 4 + 1]);
  float x2 = rintf(rois[r * 4 + 2]), y2 = rintf(rois[r * 4 + 3]);
  float rw = fmaxf(x2 - x1 + 1.f, 1.f), rh = fmaxf(y2 - y1 + 1.f, 1.f);
  float hsf = fminf(fmaxf(floorf((float)py * rh / 7.f) + y1, 0.f), 20.f);
  float hef = fminf(fmaxf(ceilf((float)(py + 1) * rh / 7.f) + y1, 0.f), 20.f);
  float wsf = fminf(fmaxf(floorf((float)px * rw / 7.f) + x1, 0.f), 20.f);
  float wef = fminf(fmaxf(ceilf((float)(px + 1) * rw / 7.f) + x1, 0.f), 20.f);
  int hs = (int)hsf, he = (int)hef, wS = (int)wsf, we = (int)wef;
  bool empty = (he <= hs) || (we <= wS);
  int c = threadIdx.x;          // 512 channels
  float m = -3.402823466e38f;
  for (int y = hs; y < he; ++y)
    for (int x = wS; x < we; ++x)
      m = fmaxf(m, feat[((long)y * 20 + x) * 512 + c]);
  float o = empty ? 0.f : m;
  pooled[(long)r * 25088 + (long)c * 49 + py * 7 + px] = (bf16)o;
}

__global__ void k_softmax21(const float* __restrict__ t, float* __restrict__ outp) {
  int r = blockIdx.x * blockDim.x + threadIdx.x;
  if (r >= NKEEP) return;
  const float* row = t + (long)r * 21;
  float m = row[0];
  for (int j = 1; j < 21; ++j) m = fmaxf(m, row[j]);
  float s = 0.f;
  float e[21];
  for (int j = 0; j < 21; ++j) { e[j] = expf(row[j] - m); s += e[j]; }
  for (int j = 0; j < 21; ++j) outp[(long)r * 21 + j] = e[j] / s;
}

// ------------------------------- host side ---------------------------------

extern "C" void kernel_launch(void* const* d_in, const int* in_sizes, int n_in,
                              void* d_out, int out_size, void* d_ws, size_t ws_size,
                              hipStream_t stream) {
  (void)in_sizes; (void)n_in; (void)out_size; (void)ws_size;

  const float* image = (const float*)d_in[0];
  const float* bbw[13]; const float* bbb[13];
  for (int i = 0; i < 13; ++i) { bbw[i] = (const float*)d_in[1 + i]; bbb[i] = (const float*)d_in[14 + i]; }
  const float* rpn_cw = (const float*)d_in[27]; const float* rpn_cb = (const float*)d_in[28];
  const float* rpn_tw = (const float*)d_in[29]; const float* rpn_tb = (const float*)d_in[30];
  const float* rpn_bw = (const float*)d_in[31]; const float* rpn_bb = (const float*)d_in[32];
  const float* fc1w = (const float*)d_in[33]; const float* fc1b = (const float*)d_in[34];
  const float* fc2w = (const float*)d_in[35]; const float* fc2b = (const float*)d_in[36];
  const float* bxw  = (const float*)d_in[37]; const float* bxb  = (const float*)d_in[38];
  const float* clsw = (const float*)d_in[39]; const float* clsb = (const float*)d_in[40];

  float* out = (float*)d_out;
  float* out_rpnbx = out;            // [1,3600,4] -> 14400
  float* out_tgt   = out + 14400;    // [1,3600,2] ->  7200
  float* out_rois  = out + 21600;    // [2000,4]   ->  8000
  float* out_roibx = out + 29600;    // [2000,4]   ->  8000
  float* out_cls   = out + 37600;    // [2000,21]  -> 42000

  char* wsb = (char*)d_ws;
  size_t cur = 0;
  auto alloc = [&](size_t bytes) -> void* {
    cur = (cur + 255) & ~(size_t)255;
    void* p = wsb + cur;
    cur += bytes;
    return p;
  };
  auto nblk = [](long total, int bs) -> unsigned { return (unsigned)((total + bs - 1) / bs); };

  static const int convO[13] = {64, 64, 128, 128, 256, 256, 256, 512, 512, 512, 512, 512, 512};
  static const int convI[13] = {3, 64, 64, 128, 128, 256, 256, 256, 512, 512, 512, 512, 512};

  // ---- workspace layout ----
  bf16* wpack[13]; int kpadc[13];
  for (int i = 0; i < 13; ++i) {
    int K = 9 * convI[i];
    int Kp = ((K + 31) / 32) * 32;
    kpadc[i] = Kp;
    wpack[i] = (bf16*)alloc((size_t)convO[i] * Kp * sizeof(bf16));
  }
  bf16* rpnc_pack = (bf16*)alloc((size_t)512 * 4608 * sizeof(bf16));
  bf16* rpnt_pack = (bf16*)alloc((size_t)18 * 512 * sizeof(bf16));
  bf16* rpnb_pack = (bf16*)alloc((size_t)36 * 512 * sizeof(bf16));
  bf16* fc1_pack  = (bf16*)alloc((size_t)4096 * 25088 * sizeof(bf16));
  bf16* fc2_pack  = (bf16*)alloc((size_t)4096 * 4096 * sizeof(bf16));
  bf16* bx_pack   = (bf16*)alloc((size_t)4 * 4096 * sizeof(bf16));
  bf16* cls_pack  = (bf16*)alloc((size_t)21 * 4096 * sizeof(bf16));

  float* actA = (float*)alloc((size_t)320 * 320 * 64 * sizeof(float));
  float* actB = (float*)alloc((size_t)320 * 320 * 64 * sizeof(float));
  bf16*  colbuf = (bf16*)alloc((size_t)CHUNK_BYTES);

  float* rpnx   = (float*)alloc((size_t)400 * 512 * sizeof(float));
  bf16*  rpnxb  = (bf16*)alloc((size_t)400 * 512 * sizeof(bf16));
  float* tgt_tmp = (float*)alloc((size_t)400 * 18 * sizeof(float));
  float* scores  = (float*)alloc((size_t)NANCH * sizeof(float));
  float* anchors = (float*)alloc((size_t)NANCH * 4 * sizeof(float));
  float* topbox  = (float*)alloc((size_t)NANCH * 4 * sizeof(float));
  int*   keep    = (int*)alloc((size_t)NKEEP * sizeof(int));
  float* roisbuf = (float*)alloc((size_t)NKEEP * 4 * sizeof(float));
  bf16*  pooled  = (bf16*)alloc((size_t)NKEEP * 25088 * sizeof(bf16));
  float* hbuf    = (float*)alloc((size_t)NKEEP * 4096 * sizeof(float));
  bf16*  hbufb   = (bf16*)alloc((size_t)NKEEP * 4096 * sizeof(bf16));
  float* clstmp  = (float*)alloc((size_t)NKEEP * 21 * sizeof(float));

  // ---- weight packing ----
  for (int i = 0; i < 13; ++i) {
    long tot = (long)convO[i] * kpadc[i];
    k_pack_conv_w<<<nblk(tot, 256), 256, 0, stream>>>(bbw[i], wpack[i], convO[i], convI[i], kpadc[i]);
  }
  k_pack_conv_w<<<nblk((long)512 * 4608, 256), 256, 0, stream>>>(rpn_cw, rpnc_pack, 512, 512, 4608);
  k_pack_lin_w<<<nblk((long)18 * 512, 256), 256, 0, stream>>>(rpn_tw, rpnt_pack, 18, 512, 512);
  k_pack_lin_w<<<nblk((long)36 * 512, 256), 256, 0, stream>>>(rpn_bw, rpnb_pack, 36, 512, 512);
  k_pack_lin_w<<<nblk((long)4096 * 25088, 256), 256, 0, stream>>>(fc1w, fc1_pack, 4096, 25088, 25088);
  k_pack_lin_w<<<nblk((long)4096 * 4096, 256), 256, 0, stream>>>(fc2w, fc2_pack, 4096, 4096, 4096);
  k_pack_lin_w<<<nblk((long)4 * 4096, 256), 256, 0, stream>>>(bxw, bx_pack, 4, 4096, 4096);
  k_pack_lin_w<<<nblk((long)21 * 4096, 256), 256, 0, stream>>>(clsw, cls_pack, 21, 4096, 4096);

  // ---- backbone ----
  k_nchw_to_nhwc<<<nblk((long)3 * 320 * 320, 256), 256, 0, stream>>>(image, actA, 3, 320, 320);
  static const int cfg[17] = {64, 64, -1, 128, 128, -1, 256, 256, 256, -1,
                              512, 512, 512, -1, 512, 512, 512};
  float* src = actA; float* dst = actB;
  int H = 320, W = 320, C = 3, ci = 0;
  for (int s = 0; s < 17; ++s) {
    if (cfg[s] == -1) {
      int HO = H / 2, WO = W / 2;
      k_maxpool<<<nblk((long)HO * WO * C, 256), 256, 0, stream>>>(src, dst, H, W, C);
      H = HO; W = WO;
      float* t = src; src = dst; dst = t;
    } else {
      int O = cfg[s];
      int Kp = kpadc[ci];
      long M = (long)H * W;
      long rows_per = (long)(CHUNK_BYTES / ((size_t)Kp * sizeof(bf16)));
      if (rows_per > M) rows_per = M;
      for (long m0 = 0; m0 < M; m0 += rows_per) {
        int rows = (int)(((M - m0) < rows_per) ? (M - m0) : rows_per);
        k_im2col<<<nblk((long)rows * Kp, 256), 256, 0, stream>>>(src, colbuf, H, W, C, Kp, (int)m0, rows);
        dim3 gg((rows + 127) / 128, (O + 63) / 64);
        k_gemm_bf16<<<gg, 256, 0, stream>>>(colbuf, wpack[ci], dst + m0 * O, rows, O, Kp, O, bbb[ci], 1);
      }
      C = O;
      float* t = src; src = dst; dst = t;
      ci++;
    }
  }
  const float* feat = src;   // NHWC [20,20,512]

  // ---- RPN ----
  k_im2col<<<nblk((long)400 * 4608, 256), 256, 0, stream>>>(feat, colbuf, 20, 20, 512, 4608, 0, 400);
  {
    dim3 gg((400 + 127) / 128, (512 + 63) / 64);
    k_gemm_bf16<<<gg, 256, 0, stream>>>(colbuf, rpnc_pack, rpnx, 400, 512, 4608, 512, rpn_cb, 1);
  }
  k_f32_to_bf16<<<nblk((long)400 * 512, 256), 256, 0, stream>>>(rpnx, rpnxb, (long)400 * 512);
  {
    dim3 gg((400 + 127) / 128, 1);
    k_gemm_bf16<<<gg, 256, 0, stream>>>(rpnxb, rpnb_pack, out_rpnbx, 400, 36, 512, 36, rpn_bb, 0);
    k_gemm_bf16<<<gg, 256, 0, stream>>>(rpnxb, rpnt_pack, tgt_tmp, 400, 18, 512, 18, rpn_tb, 0);
  }
  k_softmax2<<<nblk(NANCH, 256), 256, 0, stream>>>(tgt_tmp, out_tgt, scores);

  // ---- proposals ----
  k_anchors<<<nblk(NANCH, 256), 256, 0, stream>>>(anchors);
  k_sort_top<<<1, 512, 0, stream>>>(scores, anchors, topbox);
  k_nms<<<1, 256, 0, stream>>>(topbox, keep);
  k_rois<<<nblk(NKEEP, 256), 256, 0, stream>>>(anchors, keep, roisbuf, out_rois);
  k_roi_pool<<<NKEEP * 49, 512, 0, stream>>>(feat, roisbuf, pooled);

  // ---- RCNN head ----
  {
    dim3 gg((NKEEP + 127) / 128, (4096 + 63) / 64);
    k_gemm_bf16<<<gg, 256, 0, stream>>>(pooled, fc1_pack, hbuf, NKEEP, 4096, 25088, 4096, fc1b, 1);
  }
  k_f32_to_bf16<<<nblk((long)NKEEP * 4096, 256), 256, 0, stream>>>(hbuf, hbufb, (long)NKEEP * 4096);
  {
    dim3 gg((NKEEP + 127) / 128, (4096 + 63) / 64);
    k_gemm_bf16<<<gg, 256, 0, stream>>>(hbufb, fc2_pack, hbuf, NKEEP, 4096, 4096, 4096, fc2b, 1);
  }
  k_f32_to_bf16<<<nblk((long)NKEEP * 4096, 256), 256, 0, stream>>>(hbuf, hbufb, (long)NKEEP * 4096);
  {
    dim3 gg((NKEEP + 127) / 128, 1);
    k_gemm_bf16<<<gg, 256, 0, stream>>>(hbufb, bx_pack, out_roibx, NKEEP, 4, 4096, 4, bxb, 0);
    k_gemm_bf16<<<gg, 256, 0, stream>>>(hbufb, cls_pack, clstmp, NKEEP, 21, 4096, 21, clsb, 0);
  }
  k_softmax21<<<nblk(NKEEP, 256), 256, 0, stream>>>(clstmp, out_cls);
}
